// HMM_37924561224056
// MI455X (gfx1250) — matrix operand order repaired and checked by reading, not compiled, not run on hardware
//
#include <hip/hip_runtime.h>
#include <math.h>

typedef __attribute__((ext_vector_type(16))) _Float16 v16h;
typedef __attribute__((ext_vector_type(8)))  _Float16 v8h;
typedef __attribute__((ext_vector_type(8)))  float    v8f;

#define NEG_INF_F (-1e30f)
#define LOG2PI_F  (1.8378770664093453f)

// Problem dims (fixed by reference)
constexpr int Bn = 8, Nn = 128, Tn = 128, MELn = 80, ENCn = 512;
constexpr int PREn = 256, RNNn = 256, DECHn = 256, OUTn = 161;
constexpr int GATES = 4 * RNNn;     // 1024
constexpr int KAR   = 96;           // AR prenet K padded 80 -> 96

// ---------------------------------------------------------------------------
// Helpers
// ---------------------------------------------------------------------------
__device__ __forceinline__ float sigmoidf_(float x) { return 1.f / (1.f + expf(-x)); }
__device__ __forceinline__ float logc_sig(float x) {
  return logf(fmaxf(sigmoidf_(x), 1e-4f));
}
__device__ __forceinline__ float logaddexp_f(float a, float b) {
  float m = fmaxf(a, b), n = fminf(a, b);
  return m + log1pf(expf(n - m));
}

// ---------------------------------------------------------------------------
// Prep kernels (one-shot data layout; trivial cost)
// ---------------------------------------------------------------------------
// B matrices -> f16 col-major, K zero-padded: out[c*Rpad + r] = (r<R)?in[r*C+c]:0
__global__ __launch_bounds__(256) void transpose_pad_f16(const float* __restrict__ in,
                                                         _Float16* __restrict__ out,
                                                         int R, int C, int Rpad) {
  int idx = blockIdx.x * 256 + threadIdx.x;
  if (idx >= C * Rpad) return;
  int c = idx / Rpad, r = idx - c * Rpad;
  out[idx] = (_Float16)((r < R) ? in[(long)r * C + c] : 0.f);
}

// Materialize shifted AR prenet input: out[(b*T+t)*96 + k] = (t>0 && k<80)?mel[b,k,t-1]:0
__global__ __launch_bounds__(256) void ar_pack(const float* __restrict__ mel,
                                               float* __restrict__ out) {
  int idx = blockIdx.x * 256 + threadIdx.x;
  if (idx >= Bn * Tn * KAR) return;
  int row = idx / KAR, k = idx - row * KAR;
  int t = row & (Tn - 1), b = row >> 7;
  float v = 0.f;
  if (t > 0 && k < MELn) v = mel[(long)b * MELn * Tn + (long)k * Tn + (t - 1)];
  out[idx] = v;
}

// Interleave LSTM weights: wg[(k*256+j)*8 + g] = {w_ih | w_hh}[(g%4)*256+j][k]
// -> per (k, thread j): two contiguous float4 = two coalesced global_load_b128
__global__ __launch_bounds__(256) void pack_lstm(const float* __restrict__ w_ih,
                                                 const float* __restrict__ w_hh,
                                                 float* __restrict__ wg) {
  int idx = blockIdx.x * 256 + threadIdx.x;
  if (idx >= RNNn * RNNn * 8) return;
  int g = idx & 7, rest = idx >> 3;
  int j = rest & 255, k = rest >> 8;
  const float* src = (g < 4) ? w_ih : w_hh;
  wg[idx] = src[((long)((g & 3) * RNNn + j)) * RNNn + k];
}

// ---------------------------------------------------------------------------
// 16x16-tile WMMA GEMM: C = act(A @ B + bias); A f32 row-major (lda == kpad,
// multiple of 32), B pre-transposed f16 col-major (stride kpad).
// grid = (N/16, M/16), block = 32 (one wave per tile).
// ---------------------------------------------------------------------------
__global__ __launch_bounds__(32) void gemm_wmma_f16(
    const float* __restrict__ A,
    const _Float16* __restrict__ BT,
    const float* __restrict__ bias,
    float* __restrict__ C, int ldc, int kpad, int relu) {
  const int lane = threadIdx.x & 31;
  const int hs   = lane >> 4;
  const int l15  = lane & 15;
  const int mt   = blockIdx.y, nt = blockIdx.x;
  const int arow = mt * 16 + l15;
  const int col  = nt * 16 + l15;

  v8f acc = {};
  const int nkb = kpad >> 5;
  for (int kb = 0; kb < nkb; ++kb) {
    const int kb0 = kb * 32;
    // A fragment: 8+8 contiguous f32 -> f16  (two pairs of global_load_b128)
    const float* ap = A + (long)arow * kpad + kb0 + 8 * hs;
    float4 a0 = *reinterpret_cast<const float4*>(ap);
    float4 a1 = *reinterpret_cast<const float4*>(ap + 4);
    float4 a2 = *reinterpret_cast<const float4*>(ap + 16);
    float4 a3 = *reinterpret_cast<const float4*>(ap + 20);
    v16h af;
    af[0]=(_Float16)a0.x; af[1]=(_Float16)a0.y; af[2]=(_Float16)a0.z; af[3]=(_Float16)a0.w;
    af[4]=(_Float16)a1.x; af[5]=(_Float16)a1.y; af[6]=(_Float16)a1.z; af[7]=(_Float16)a1.w;
    af[8]=(_Float16)a2.x; af[9]=(_Float16)a2.y; af[10]=(_Float16)a2.z; af[11]=(_Float16)a2.w;
    af[12]=(_Float16)a3.x; af[13]=(_Float16)a3.y; af[14]=(_Float16)a3.z; af[15]=(_Float16)a3.w;
    // B fragment: 8+8 f16 (k = 8h..8h+7, 16+8h..) from col-major BT
    const _Float16* bp = BT + (long)col * kpad + kb0 + 8 * hs;
    v8h b0 = *reinterpret_cast<const v8h*>(bp);
    v8h b1 = *reinterpret_cast<const v8h*>(bp + 16);
    v16h bf = __builtin_shufflevector(b0, b1, 0,1,2,3,4,5,6,7,8,9,10,11,12,13,14,15);
    acc = __builtin_amdgcn_wmma_f32_16x16x32_f16(false, af, false, bf,
                                                 (short)0, acc, false, false);
  }
  const float bv = bias ? bias[col] : 0.f;
#pragma unroll
  for (int r = 0; r < 8; ++r) {
    int row = mt * 16 + r + 8 * hs;
    float v = acc[r] + bv;
    if (relu) v = fmaxf(v, 0.f);
    C[(long)row * ldc + col] = v;
  }
}

// ---------------------------------------------------------------------------
// Persistent scan kernel: one workgroup per batch element, T=128 steps inside.
// ---------------------------------------------------------------------------
constexpr int F32_COUNT =
    Nn * DECHn + OUTn + MELn + PREn + RNNn + RNNn + DECHn + Nn * 5 + 32;
constexpr size_t F32_PAD    = ((size_t)F32_COUNT * 4 + 15) / 16 * 16;
constexpr size_t SMEM_BYTES = F32_PAD + (size_t)Nn * DECHn * 2 + (size_t)OUTn * DECHn * 2;
static_assert(SMEM_BYTES <= 320 * 1024, "LDS budget exceeded");

__global__ __launch_bounds__(256) void scan_kernel(
    const int*   __restrict__ text_len,
    const float* __restrict__ mel_in,       // (B, MEL, T)
    const int*   __restrict__ mel_len,
    const float* __restrict__ wg,           // packed LSTM weights (k,j,8)
    const float* __restrict__ b_ih,
    const float* __restrict__ b_hh,
    const float* __restrict__ dec_w1,       // (768, 256); rows 0..255 = h part
    const float* __restrict__ dec_w2,       // (256, 161)
    const float* __restrict__ dec_b2,       // (161)
    const float* __restrict__ ws_te,        // (B*N, 256) hoisted text proj (+b1)
    const float* __restrict__ ws_pre,       // (B*T, 256) prenet output
    float*       __restrict__ out) {
  extern __shared__ char smem_raw[];
  float* te     = reinterpret_cast<float*>(smem_raw);   // 128*256
  float* b2     = te + Nn * DECHn;                      // 161
  float* xt     = b2 + OUTn;                            // 80
  float* pre_s  = xt + MELn;                            // 256
  float* h      = pre_s + PREn;                         // 256
  float* c      = h + RNNn;                             // 256
  float* hp     = c + RNNn;                             // 256
  float* em     = hp + DECHn;                           // 128
  float* tv     = em + Nn;                              // 128
  float* la     = tv + Nn;                              // 128
  float* lastla = la + Nn;                              // 128
  float* lasttv = lastla + Nn;                          // 128
  float* red    = lasttv + Nn;                          // 32
  _Float16* dh  = reinterpret_cast<_Float16*>(smem_raw + F32_PAD);  // 128x256 row-major
  _Float16* w2c = dh + Nn * DECHn;                      // 161x256 col-major

  const int b    = blockIdx.x;
  const int tid  = threadIdx.x;
  const int lane = tid & 31;
  const int wv   = tid >> 5;
  const int tlen = text_len[b];
  const int mlen = mel_len[b];

  for (int i = tid; i < Nn * DECHn; i += 256) te[i] = ws_te[(long)b * Nn * DECHn + i];
  for (int i = tid; i < OUTn * DECHn; i += 256) {
    int col = i >> 8, k = i & 255;
    w2c[i] = (_Float16)dec_w2[(long)k * OUTn + col];
  }
  if (tid < OUTn) b2[tid] = dec_b2[tid];
  h[tid] = 0.f; c[tid] = 0.f;
  if (tid < Nn) la[tid] = 0.f;
  const float gb0 = b_ih[tid]       + b_hh[tid];
  const float gb1 = b_ih[256 + tid] + b_hh[256 + tid];
  const float gb2 = b_ih[512 + tid] + b_hh[512 + tid];
  const float gb3 = b_ih[768 + tid] + b_hh[768 + tid];
  float logc_sum = 0.f;
  __syncthreads();

  const int hs  = lane >> 4;
  const int l15 = lane & 15;

  for (int t = 0; t < Tn; ++t) {
    if (tid < MELn) xt[tid] = mel_in[(long)b * MELn * Tn + tid * Tn + t];
    pre_s[tid] = ws_pre[((long)b * Tn + t) * PREn + tid];
    __syncthreads();

    // ---- LSTM gates: 2 coalesced b128 loads + 8 FMA per k ----
    float g0 = gb0, g1 = gb1, g2 = gb2, g3 = gb3;
#pragma unroll 2
    for (int k = 0; k < RNNn; ++k) {
      float p = pre_s[k], hh = h[k];
      const float4* w4 = reinterpret_cast<const float4*>(wg + (((long)k * RNNn + tid) << 3));
      float4 wi = w4[0], wh = w4[1];
      g0 += p * wi.x + hh * wh.x;
      g1 += p * wi.y + hh * wh.y;
      g2 += p * wi.z + hh * wh.z;
      g3 += p * wi.w + hh * wh.w;
    }
    __syncthreads();
    {
      float cn = sigmoidf_(g1) * c[tid] + sigmoidf_(g0) * tanhf(g2);
      c[tid] = cn;
      h[tid] = sigmoidf_(g3) * tanhf(cn);
    }
    __syncthreads();

    // ---- h projection through dec_w1 rows 0..255 (coalesced b32) ----
    {
      float acc = 0.f;
      for (int k = 0; k < RNNn; ++k) acc += h[k] * dec_w1[(long)k * DECHn + tid];
      hp[tid] = acc;
    }
    __syncthreads();

    // ---- dh = relu(te_proj + h_proj) -> f16 in LDS ----
    for (int i = tid; i < Nn * DECHn; i += 256)
      dh[i] = (_Float16)fmaxf(te[i] + hp[i & 255], 0.f);
    __syncthreads();

    // ---- WMMA: out = dh(128x256) @ w2(256x160); wave wv owns rows 16wv..+15 ----
    v16h afr[8];
    {
      const int rrow = 16 * wv + l15;
#pragma unroll
      for (int kb = 0; kb < 8; ++kb) {
        int base = kb * 32 + hs * 8;
        v8h lo = *reinterpret_cast<const v8h*>(&dh[rrow * 256 + base]);
        v8h hi = *reinterpret_cast<const v8h*>(&dh[rrow * 256 + base + 16]);
        afr[kb] = __builtin_shufflevector(lo, hi, 0,1,2,3,4,5,6,7,8,9,10,11,12,13,14,15);
      }
    }
    float em_acc[8];
#pragma unroll
    for (int r = 0; r < 8; ++r) em_acc[r] = 0.f;
#pragma unroll
    for (int j = 0; j < 5; ++j) {           // mean tile j pairs with std tile 5+j
      v8f am = {}, as = {};
      const int cm = 16 * j + l15;          // mean column
      const int cs = 80 + cm;               // std column
#pragma unroll
      for (int kb = 0; kb < 8; ++kb) {
        int kbase = kb * 32 + 8 * hs;
        v8h m0 = *reinterpret_cast<const v8h*>(&w2c[cm * 256 + kbase]);
        v8h m1 = *reinterpret_cast<const v8h*>(&w2c[cm * 256 + kbase + 16]);
        v16h bfm = __builtin_shufflevector(m0, m1, 0,1,2,3,4,5,6,7,8,9,10,11,12,13,14,15);
        am = __builtin_amdgcn_wmma_f32_16x16x32_f16(false, afr[kb], false, bfm,
                                                    (short)0, am, false, false);
        v8h s0 = *reinterpret_cast<const v8h*>(&w2c[cs * 256 + kbase]);
        v8h s1 = *reinterpret_cast<const v8h*>(&w2c[cs * 256 + kbase + 16]);
        v16h bfs = __builtin_shufflevector(s0, s1, 0,1,2,3,4,5,6,7,8,9,10,11,12,13,14,15);
        as = __builtin_amdgcn_wmma_f32_16x16x32_f16(false, afr[kb], false, bfs,
                                                    (short)0, as, false, false);
      }
#pragma unroll
      for (int r = 0; r < 8; ++r) {
        float mean = am[r] + b2[cm];
        float sv   = as[r] + b2[cs];
        float sp   = (sv > 20.f) ? sv : log1pf(expf(sv));
        float stdv = fmaxf(sp, 0.001f);
        float d    = (xt[cm] - mean) / stdv;
        em_acc[r] += -0.5f * d * d - logf(stdv);
      }
    }
#pragma unroll
    for (int r = 0; r < 8; ++r) {
      float v = em_acc[r];
      v += __shfl_xor(v, 1); v += __shfl_xor(v, 2);
      v += __shfl_xor(v, 4); v += __shfl_xor(v, 8);
      if (l15 == 0) em[16 * wv + r + 8 * hs] = v - 0.5f * LOG2PI_F * (float)MELn;
    }
    // tv = dh @ w2[:,160] + b2[160]  (vectorized v8h LDS reads)
    if (tid < Nn) {
      float acc = b2[160];
      const v8h* dv = reinterpret_cast<const v8h*>(&dh[tid * 256]);
      const v8h* wc = reinterpret_cast<const v8h*>(&w2c[160 * 256]);
      for (int k8 = 0; k8 < 32; ++k8) {
        v8h a = dv[k8], w = wc[k8];
#pragma unroll
        for (int e = 0; e < 8; ++e) acc += (float)a[e] * (float)w[e];
      }
      tv[tid] = acc;
    }
    __syncthreads();

    // ---- HMM forward recursion ----
    float lat_v = NEG_INF_F;
    if (tid < Nn) {
      int n = tid;
      float em_n = (n < tlen) ? em[n] : 0.f;
      if (t == 0) {
        lat_v = ((n == 0) ? 0.f : NEG_INF_F) + em_n;
      } else {
        float staying = la[n] + logc_sig(-tv[n]);
        float leaving = (n == 0) ? NEG_INF_F : la[n - 1] + logc_sig(tv[n - 1]);
        float trans   = (n < tlen) ? logaddexp_f(staying, leaving) : NEG_INF_F;
        lat_v = em_n + trans;
      }
    }
    // block logsumexp over 128 states
    float m = lat_v;
    m = fmaxf(m, __shfl_xor(m, 16)); m = fmaxf(m, __shfl_xor(m, 8));
    m = fmaxf(m, __shfl_xor(m, 4));  m = fmaxf(m, __shfl_xor(m, 2));
    m = fmaxf(m, __shfl_xor(m, 1));
    if (lane == 0) red[wv] = m;
    __syncthreads();
    float gmax = red[0];
    for (int w = 1; w < 8; ++w) gmax = fmaxf(gmax, red[w]);
    __syncthreads();
    float ex = (tid < Nn) ? expf(lat_v - gmax) : 0.f;
    ex += __shfl_xor(ex, 16); ex += __shfl_xor(ex, 8);
    ex += __shfl_xor(ex, 4);  ex += __shfl_xor(ex, 2);
    ex += __shfl_xor(ex, 1);
    if (lane == 0) red[wv] = ex;
    __syncthreads();
    float gsum = 0.f;
    for (int w = 0; w < 8; ++w) gsum += red[w];
    float lse = gmax + logf(gsum);
    if (t < mlen) logc_sum += lse;
    __syncthreads();
    if (tid < Nn) {
      float lan = lat_v - lse;
      la[tid] = lan;
      if (t == mlen - 1) {
        lastla[tid] = (tid < tlen) ? lan : NEG_INF_F;
        lasttv[tid] = tv[tid];
      }
    }
    __syncthreads();
  }

  if (tid == 0) {
    int n = tlen - 1;
    out[b] = logc_sum + lastla[n] + logc_sig(lasttv[n]);
  }
}

// ---------------------------------------------------------------------------
// Host launcher
// ---------------------------------------------------------------------------
extern "C" void kernel_launch(void* const* d_in, const int* in_sizes, int n_in,
                              void* d_out, int out_size, void* d_ws, size_t ws_size,
                              hipStream_t stream) {
  (void)in_sizes; (void)n_in; (void)out_size; (void)ws_size;
  const float* text_emb  = (const float*)d_in[0];
  const int*   text_len  = (const int*)d_in[1];
  const float* mel_in    = (const float*)d_in[2];
  const int*   mel_len   = (const int*)d_in[3];
  const float* prenet_w1 = (const float*)d_in[4];
  const float* prenet_w2 = (const float*)d_in[5];
  const float* w_ih      = (const float*)d_in[6];
  const float* w_hh      = (const float*)d_in[7];
  const float* b_ih      = (const float*)d_in[8];
  const float* b_hh      = (const float*)d_in[9];
  const float* dec_w1    = (const float*)d_in[10];
  const float* dec_b1    = (const float*)d_in[11];
  const float* dec_w2    = (const float*)d_in[12];
  const float* dec_b2    = (const float*)d_in[13];
  float* out = (float*)d_out;

  // ---- workspace carve-up ----
  char* cur = (char*)d_ws;
  auto takeF = [&](size_t n) { float* p = (float*)cur; cur += n * 4; return p; };
  auto takeH = [&](size_t n) { _Float16* p = (_Float16*)cur; cur += ((n * 2 + 15) / 16) * 16; return p; };
  float*    ws_te  = takeF((size_t)Bn * Nn * DECHn);   // 1 MB
  float*    ws_ar  = takeF((size_t)Bn * Tn * KAR);     // 384 KB shifted AR input
  float*    ws_p1  = takeF((size_t)Bn * Tn * PREn);    // 1 MB
  float*    ws_pre = takeF((size_t)Bn * Tn * PREn);    // 1 MB
  float*    wg     = takeF((size_t)RNNn * RNNn * 8);   // 2 MB packed LSTM weights
  _Float16* dw1T   = takeH((size_t)DECHn * ENCn);      // 256 KB [col*512+k]
  _Float16* pw2T   = takeH((size_t)PREn * PREn);       // 128 KB [col*256+k]
  _Float16* pw1T   = takeH((size_t)PREn * KAR);        // 48 KB  [col*96+k]

  // ---- prep: one-shot data layout ----
  transpose_pad_f16<<<(DECHn * ENCn + 255) / 256, 256, 0, stream>>>(
      dec_w1 + RNNn * DECHn, dw1T, ENCn, DECHn, ENCn);
  transpose_pad_f16<<<(PREn * PREn + 255) / 256, 256, 0, stream>>>(
      prenet_w2, pw2T, PREn, PREn, PREn);
  transpose_pad_f16<<<(PREn * KAR + 255) / 256, 256, 0, stream>>>(
      prenet_w1, pw1T, MELn, PREn, KAR);
  ar_pack<<<(Bn * Tn * KAR + 255) / 256, 256, 0, stream>>>(mel_in, ws_ar);
  pack_lstm<<<(RNNn * RNNn * 8 + 255) / 256, 256, 0, stream>>>(w_ih, w_hh, wg);

  dim3 tgrid(DECHn / 16, (Bn * Nn) / 16);  // (16, 64)
  // 1) hoisted text projection: te = text_emb @ dec_w1[256:,:] + dec_b1
  gemm_wmma_f16<<<tgrid, 32, 0, stream>>>(text_emb, dw1T, dec_b1,
                                          ws_te, DECHn, ENCn, /*relu*/0);
  // 2) prenet layer 1: relu(ar @ prenet_w1)
  gemm_wmma_f16<<<tgrid, 32, 0, stream>>>(ws_ar, pw1T, nullptr,
                                          ws_p1, PREn, KAR, /*relu*/1);
  // 3) prenet layer 2: relu(p1 @ prenet_w2)  (A re-padded? no: kpad==256 exact)
  gemm_wmma_f16<<<tgrid, 32, 0, stream>>>(ws_p1, pw2T, nullptr,
                                          ws_pre, PREn, PREn, /*relu*/1);

  // 4) persistent per-batch scan (dynamic LDS ~280 KB of the 320 KB WGP pool)
  (void)hipFuncSetAttribute(reinterpret_cast<const void*>(&scan_kernel),
                            hipFuncAttributeMaxDynamicSharedMemorySize,
                            (int)SMEM_BYTES);
  scan_kernel<<<Bn, 256, SMEM_BYTES, stream>>>(text_len, mel_in, mel_len,
                                               wg, b_ih, b_hh,
                                               dec_w1, dec_w2, dec_b2,
                                               ws_te, ws_pre, out);
}
